// SwinTransformerBlock_76493367542590
// MI455X (gfx1250) — matrix operand order repaired and checked by reading, not compiled, not run on hardware
//
#include <hip/hip_runtime.h>
#include <hip/hip_bf16.h>

// ---------------------------------------------------------------------------
// Swin-3D block for gfx1250 (MI455X). All GEMMs on v_wmma_f32_16x16x32_f16.
// wave32; 256-thread workgroups (8 waves). LDS tiles staged with the
// Tensor Data Mover (tensor_load_to_lds + s_wait_tensorcnt) when available.
// ---------------------------------------------------------------------------

typedef _Float16 half_t;
typedef __attribute__((ext_vector_type(16))) _Float16 v16h;
typedef __attribute__((ext_vector_type(8)))  float    v8f;
typedef unsigned int u32x4 __attribute__((ext_vector_type(4)));
typedef int          i32x4 __attribute__((ext_vector_type(4)));
typedef int          i32x8 __attribute__((ext_vector_type(8)));

#if __has_builtin(__builtin_amdgcn_tensor_load_to_lds)
#define USE_TDM 1
#else
#define USE_TDM 0
#endif

__device__ __forceinline__ v8f wmma32(v16h a, v16h b, v8f c) {
  // (neg_a, A, neg_b, B, c_mod, C, reuse_a, reuse_b)
  return __builtin_amdgcn_wmma_f32_16x16x32_f16(false, a, false, b, (short)0, c,
                                                false, false);
}

__device__ __forceinline__ void wait_tensorcnt0() {
#if __has_builtin(__builtin_amdgcn_s_wait_tensorcnt)
  __builtin_amdgcn_s_wait_tensorcnt(0);
#else
  asm volatile("s_wait_tensorcnt 0x0" ::: "memory");
#endif
}

#if USE_TDM
// TDM: load a 2D f16 tile (tile_w x tile_h elements, row stride stride_elems)
// from global into LDS at byte offset lds_off. LDS rows are padded:
// pad_interval=3 -> pad every 16 DWORDs (64B row), pad_amount=3 -> 4 DWORDs
// (16B), matching a 40-half LDS row stride for a 32-half tile row.
__device__ __forceinline__ void tdm_load_tile_f16(unsigned lds_off,
                                                  const half_t* gptr,
                                                  unsigned tile_w,
                                                  unsigned tile_h,
                                                  unsigned stride_elems) {
  unsigned long long ga = (unsigned long long)(__UINTPTR_TYPE__)gptr;
  u32x4 g0;
  g0[0] = 1u;                                        // count=1, user descriptor
  g0[1] = lds_off;                                   // LDS byte address
  g0[2] = (unsigned)(ga & 0xffffffffu);              // global addr [31:0]
  g0[3] = (unsigned)((ga >> 32) & 0x01ffffffu)       // global addr [56:32]
        | (2u << 30);                                // type = 2 ("image")
  i32x8 g1;
  g1[0] = (int)((1u << 16)                           // data_size = 2 bytes
              | (1u << 20)                           // pad_enable
              | (3u << 22)                           // pad_interval: 16 DWORDs
              | (3u << 25));                         // pad_amount: 4 DWORDs
  g1[1] = (int)((tile_w & 0xffffu) << 16);           // tensor_dim0 [15:0]
  g1[2] = (int)(((tile_w >> 16) & 0xffffu)           // tensor_dim0 [31:16]
              | ((tile_h & 0xffffu) << 16));         // tensor_dim1 [15:0]
  g1[3] = (int)(((tile_h >> 16) & 0xffffu)           // tensor_dim1 [31:16]
              | ((tile_w & 0xffffu) << 16));         // tile_dim0
  g1[4] = (int)(tile_h & 0xffffu);                   // tile_dim1 (tile_dim2=0)
  g1[5] = (int)stride_elems;                         // tensor_dim0_stride lo
  g1[6] = 0;                                         // stride hi / dim1_stride
  g1[7] = 0;
  i32x4 z4 = {};
  i32x8 z8 = {};
  // 6-arg form (clang-23 / therock): (g0, g1, g2, g3, g4, cpol)
  __builtin_amdgcn_tensor_load_to_lds(g0, g1, z4, z4, z8, 0);
}
#endif

// ---------------------------------------------------------------------------
// fp32 -> fp16 weight conversion
// ---------------------------------------------------------------------------
__global__ __launch_bounds__(256) void f32_to_f16_kernel(
    const float* __restrict__ in, half_t* __restrict__ out, int n) {
  int i = blockIdx.x * 256 + threadIdx.x;
  if (i < n) out[i] = (half_t)in[i];
}

// ---------------------------------------------------------------------------
// Expand relative-position-bias table -> bias[h][n][m], n,m in [0,256)
// window (4,8,8): rel index = (dd+3)*225 + (dh+7)*15 + (dw+7)
// ---------------------------------------------------------------------------
__global__ __launch_bounds__(256) void bias_expand_kernel(
    const float* __restrict__ tbl, float* __restrict__ bias) {
  int e = blockIdx.x * 256 + threadIdx.x;      // 12*256*256 = 786432
  int h  = e >> 16;
  int nm = e & 65535;
  int n = nm >> 8, m = nm & 255;
  int nd = n >> 6, nh = (n >> 3) & 7, nw = n & 7;
  int md = m >> 6, mh = (m >> 3) & 7, mw = m & 7;
  int idx = (nd - md + 3) * 225 + (nh - mh + 7) * 15 + (nw - mw + 7);
  bias[e] = tbl[idx * 12 + h];
}

// ---------------------------------------------------------------------------
// LayerNorm (C=384) producing fp16 rows.
// mode 0: read x with cyclic-shift + window-partition mapping (LN1)
// mode 1: identity token order (LN2)
// One token per 32 lanes, 8 tokens per 256-thread block.
// ---------------------------------------------------------------------------
__global__ __launch_bounds__(256) void ln_kernel(
    const float* __restrict__ x, const float* __restrict__ gw,
    const float* __restrict__ gb, half_t* __restrict__ out, int mode) {
  int tok  = blockIdx.x * 8 + (threadIdx.x >> 5);
  int lane = threadIdx.x & 31;
  const float* src;
  if (mode == 0) {
    int w = tok >> 8, t = tok & 255;
    int dblk = w >> 6, hb = (w >> 3) & 7, wb = w & 7;
    int td = t >> 6, th = (t >> 3) & 7, tw = t & 7;
    int d  = (dblk * 4 + td + 2) & 7;     // shifted coord + ss, mod dims
    int hh = (hb * 8 + th + 4) & 63;
    int ww = (wb * 8 + tw + 4) & 63;
    src = x + (((size_t)d * 64 + hh) * 64 + ww) * 384;
  } else {
    src = x + (size_t)tok * 384;
  }
  half_t* dst = out + (size_t)tok * 384;

  float v[12];
  float s = 0.f;
#pragma unroll
  for (int j = 0; j < 12; ++j) { v[j] = src[lane + 32 * j]; s += v[j]; }
#pragma unroll
  for (int m = 16; m >= 1; m >>= 1) s += __shfl_xor(s, m, 32);
  float mean = s * (1.0f / 384.0f);
  float q = 0.f;
#pragma unroll
  for (int j = 0; j < 12; ++j) { float d2 = v[j] - mean; q += d2 * d2; }
#pragma unroll
  for (int m = 16; m >= 1; m >>= 1) q += __shfl_xor(q, m, 32);
  float rstd = rsqrtf(q * (1.0f / 384.0f) + 1e-5f);
#pragma unroll
  for (int j = 0; j < 12; ++j) {
    int c = lane + 32 * j;
    dst[c] = (half_t)((v[j] - mean) * rstd * gw[c] + gb[c]);
  }
}

// ---------------------------------------------------------------------------
// Generic WMMA GEMM: OUT[M][N] = act[M][K] @ wmat[N][K]^T + bias  (+ epilogue)
// Block tile 128x64x32, 8 waves each owning 32x32 (2x2 WMMA tiles).
// EPI 0: scatter into q(scaled)/k/vT packed attention layouts
// EPI 1: proj -> window-reverse + unshift + residual add, fp32 out
// EPI 2: exact GELU, fp16 row-major out
// EPI 3: out += result (fp32, row-major)
// ---------------------------------------------------------------------------
template <int EPI>
__global__ __launch_bounds__(256) void wgemm_kernel(
    const half_t* __restrict__ act, const half_t* __restrict__ wmat,
    const float* __restrict__ bias, int M, int N, int K,
    float* __restrict__ outf, half_t* __restrict__ outh,
    half_t* __restrict__ outh2, half_t* __restrict__ outh3,
    const float* __restrict__ resid) {
  __shared__ __align__(16) half_t sA[128 * 40];   // 128x32, stride 40
  __shared__ __align__(16) half_t sB[64 * 40];    // 64x32,  stride 40

  int tid    = threadIdx.x;
  int lane   = tid & 31;
  int waveId = tid >> 5;
  int wm0 = (waveId >> 1) * 32;
  int wn0 = (waveId & 1) * 32;
  size_t m0 = (size_t)blockIdx.y * 128;
  int    n0 = blockIdx.x * 64;

  v8f acc[2][2] = {};

  for (int k0 = 0; k0 < K; k0 += 32) {
    if (k0 + 32 < K) {
      __builtin_prefetch(act + (m0 + (tid >> 1)) * K + k0 + 32, 0, 0);
      if (tid < 64)
        __builtin_prefetch(wmat + (size_t)(n0 + tid) * K + k0 + 32, 0, 0);
    }
#if USE_TDM
    // Tensor Data Mover stages both tiles; wave 0 issues, waits TENSORcnt,
    // then the workgroup barrier publishes the LDS data to all waves.
    if (waveId == 0) {
      tdm_load_tile_f16((unsigned)(__UINTPTR_TYPE__)&sA[0],
                        act + m0 * K + k0, 32, 128, (unsigned)K);
      tdm_load_tile_f16((unsigned)(__UINTPTR_TYPE__)&sB[0],
                        wmat + (size_t)n0 * K + k0, 32, 64, (unsigned)K);
      wait_tensorcnt0();
    }
#else
#pragma unroll
    for (int i = tid; i < 512; i += 256) {
      int r = i >> 2, c = (i & 3) << 3;
      *(uint4*)&sA[r * 40 + c] = *(const uint4*)(act + (m0 + r) * K + k0 + c);
    }
    {
      int r = tid >> 2, c = (tid & 3) << 3;
      *(uint4*)&sB[r * 40 + c] =
          *(const uint4*)(wmat + (size_t)(n0 + r) * K + k0 + c);
    }
#endif
    __syncthreads();

    v16h afrag[2], bfrag[2];
    int kh = (lane >> 4) << 3;
#pragma unroll
    for (int tm = 0; tm < 2; ++tm) {
      int row = wm0 + tm * 16 + (lane & 15);
      unsigned int* ap = (unsigned int*)&afrag[tm];
#pragma unroll
      for (int j = 0; j < 8; ++j) {
        int kk = ((j >> 2) << 4) + kh + ((j & 3) << 1);
        ap[j] = *(const unsigned int*)&sA[row * 40 + kk];
      }
    }
#pragma unroll
    for (int tn = 0; tn < 2; ++tn) {
      int col = wn0 + tn * 16 + (lane & 15);
      const half_t* src = &sB[col * 40 + ((lane >> 4) << 3)];
      uint4* bp = (uint4*)&bfrag[tn];
      bp[0] = *(const uint4*)src;
      bp[1] = *(const uint4*)(src + 16);
    }
#pragma unroll
    for (int tm = 0; tm < 2; ++tm)
#pragma unroll
      for (int tn = 0; tn < 2; ++tn)
        acc[tm][tn] = wmma32(afrag[tm], bfrag[tn], acc[tm][tn]);
    __syncthreads();
  }

  // ---- epilogue ----
#pragma unroll
  for (int tm = 0; tm < 2; ++tm) {
#pragma unroll
    for (int tn = 0; tn < 2; ++tn) {
#pragma unroll
      for (int r = 0; r < 8; ++r) {
        int mrow = (int)m0 + wm0 + tm * 16 + ((lane >> 4) << 3) + r;
        int ncol = n0 + wn0 + tn * 16 + (lane & 15);
        float vv = acc[tm][tn][r] + bias[ncol];
        if (EPI == 0) {
          // qkv scatter: w=window, t=token; sec 0=q (scaled), 1=k, 2=vT
          int w = mrow >> 8, t = mrow & 255;
          int sec = ncol / 384;
          int rem = ncol - sec * 384;
          int hh = rem >> 5, hd = rem & 31;
          size_t base = (size_t)(w * 12 + hh);
          if (sec == 0)
            outh[(base * 256 + t) * 32 + hd] = (half_t)(vv * 0.17677669529663687f);
          else if (sec == 1)
            outh2[(base * 256 + t) * 32 + hd] = (half_t)vv;
          else
            outh3[(base * 32 + hd) * 256 + t] = (half_t)vv;
        } else if (EPI == 1) {
          // proj: window-reverse + reverse cyclic shift + residual
          int w = mrow >> 8, t = mrow & 255;
          int dblk = w >> 6, hb = (w >> 3) & 7, wb = w & 7;
          int td = t >> 6, th = (t >> 3) & 7, tw = t & 7;
          int d  = (dblk * 4 + td + 2) & 7;
          int hh = (hb * 8 + th + 4) & 63;
          int ww = (wb * 8 + tw + 4) & 63;
          size_t p = (((size_t)d * 64 + hh) * 64 + ww) * 384 + ncol;
          outf[p] = resid[p] + vv;
        } else if (EPI == 2) {
          float g = 0.5f * vv * (1.0f + erff(vv * 0.7071067811865476f));
          outh[(size_t)mrow * N + ncol] = (half_t)g;
        } else {
          outf[(size_t)mrow * N + ncol] += vv;
        }
      }
    }
  }
}

// ---------------------------------------------------------------------------
// Windowed attention: one workgroup per (window, head, 128-row q block).
// S = qK^T (+bias +shift-mask), row softmax, O = P V. All matmuls WMMA.
// ---------------------------------------------------------------------------
__global__ __launch_bounds__(256) void attn_kernel(
    const half_t* __restrict__ qp, const half_t* __restrict__ kp,
    const half_t* __restrict__ vtp, const float* __restrict__ bias,
    half_t* __restrict__ o) {
  __shared__ __align__(16) unsigned char smem[49408];
  half_t* sK = (half_t*)smem;                    // 256 x 32, stride 40 (20480B)
  half_t* sQ = (half_t*)(smem + 20480);          // 128 x 32, stride 40 (10240B)
  half_t* sP = (half_t*)(smem + 30720);          // 8 waves x 16 x 72  (18432B)
  unsigned char* sLab = smem + 49152;            // 256 region labels

  int qb   = blockIdx.x & 1;
  int wh   = blockIdx.x >> 1;      // win*12 + head
  int win  = wh / 12;
  int head = wh - win * 12;
  int tid    = threadIdx.x;
  int lane   = tid & 31;
  int waveId = tid >> 5;

  const half_t* kbase = kp + (size_t)wh * 256 * 32;
  const half_t* qbase = qp + (size_t)wh * 256 * 32 + (size_t)qb * 128 * 32;

#if USE_TDM
  if (waveId == 0) {
    tdm_load_tile_f16((unsigned)(__UINTPTR_TYPE__)sK, kbase, 32, 256, 32);
    tdm_load_tile_f16((unsigned)(__UINTPTR_TYPE__)sQ, qbase, 32, 128, 32);
    wait_tensorcnt0();
  }
#else
#pragma unroll
  for (int i = tid; i < 1024; i += 256) {        // K: 256 rows of 32 halves
    int r = i >> 2, c = (i & 3) << 3;
    *(uint4*)&sK[r * 40 + c] = *(const uint4*)(kbase + r * 32 + c);
  }
#pragma unroll
  for (int i = tid; i < 512; i += 256) {         // Q block: 128 rows
    int r = i >> 2, c = (i & 3) << 3;
    *(uint4*)&sQ[r * 40 + c] = *(const uint4*)(qbase + r * 32 + c);
  }
#endif
  {
    // shift-mask region label for token tid in this window (unshifted grid)
    int t = tid;
    int dblk = win >> 6, hb = (win >> 3) & 7, wb = win & 7;
    int d = dblk * 4 + (t >> 6);
    int h = hb * 8 + ((t >> 3) & 7);
    int w = wb * 8 + (t & 7);
    int rd = (d < 4) ? 0 : ((d < 6) ? 1 : 2);
    int rh = (h < 56) ? 0 : ((h < 60) ? 1 : 2);
    int rw = (w < 56) ? 0 : ((w < 60) ? 1 : 2);
    sLab[t] = (unsigned char)(rd * 9 + rh * 3 + rw);
  }
  __syncthreads();

  // ---- S = q @ k^T : each wave does 16 q-rows x 256 k-cols (16 tiles) ----
  v8f accS[16] = {};
  {
    v16h aq;
    unsigned int* ap = (unsigned int*)&aq;
    int row = waveId * 16 + (lane & 15);
    int kh = (lane >> 4) << 3;
#pragma unroll
    for (int j = 0; j < 8; ++j) {
      int kk = ((j >> 2) << 4) + kh + ((j & 3) << 1);
      ap[j] = *(const unsigned int*)&sQ[row * 40 + kk];
    }
#pragma unroll
    for (int tn = 0; tn < 16; ++tn) {
      const half_t* src = &sK[(tn * 16 + (lane & 15)) * 40 + ((lane >> 4) << 3)];
      v16h bk;
      uint4* bp = (uint4*)&bk;
      bp[0] = *(const uint4*)src;
      bp[1] = *(const uint4*)(src + 16);
      accS[tn] = wmma32(aq, bk, accS[tn]);
    }
  }

  // ---- bias + mask, then row softmax ----
  int nbase = qb * 128 + waveId * 16 + ((lane >> 4) << 3);
  const float* bh = bias + (size_t)head * 65536;
#pragma unroll
  for (int tn = 0; tn < 16; ++tn) {
    int mcol = tn * 16 + (lane & 15);
    int lm = sLab[mcol];
#pragma unroll
    for (int r = 0; r < 8; ++r) {
      int n = nbase + r;
      float bm = bh[(size_t)n * 256 + mcol];
      if (sLab[n] != lm) bm -= 100.0f;
      accS[tn][r] += bm;
    }
  }
#pragma unroll
  for (int r = 0; r < 8; ++r) {
    float mx = -1e30f;
#pragma unroll
    for (int tn = 0; tn < 16; ++tn) mx = fmaxf(mx, accS[tn][r]);
#pragma unroll
    for (int m = 8; m >= 1; m >>= 1) mx = fmaxf(mx, __shfl_xor(mx, m, 32));
    float sum = 0.f;
#pragma unroll
    for (int tn = 0; tn < 16; ++tn) {
      float e = __expf(accS[tn][r] - mx);
      accS[tn][r] = e;
      sum += e;
    }
#pragma unroll
    for (int m = 8; m >= 1; m >>= 1) sum += __shfl_xor(sum, m, 32);
    float inv = 1.0f / sum;
#pragma unroll
    for (int tn = 0; tn < 16; ++tn) accS[tn][r] *= inv;
  }

  // ---- O = P @ V : stream P through per-wave LDS chunks (64 cols) ----
  v8f accO[2] = {};
  half_t* pw = sP + waveId * (16 * 72);
  for (int chunk = 0; chunk < 4; ++chunk) {
#pragma unroll
    for (int tt = 0; tt < 4; ++tt) {
      int tn = chunk * 4 + tt;
#pragma unroll
      for (int r = 0; r < 8; ++r)
        pw[(((lane >> 4) << 3) + r) * 72 + tt * 16 + (lane & 15)] =
            (half_t)accS[tn][r];
    }
    __syncthreads();
#pragma unroll
    for (int ks = 0; ks < 2; ++ks) {
      int kb = chunk * 64 + ks * 32;
      v16h a;
      unsigned int* ap = (unsigned int*)&a;
      int row = lane & 15;
      int kh = (lane >> 4) << 3;
#pragma unroll
      for (int j = 0; j < 8; ++j) {
        int kk = ks * 32 + ((j >> 2) << 4) + kh + ((j & 3) << 1);
        ap[j] = *(const unsigned int*)&pw[row * 72 + kk];
      }
#pragma unroll
      for (int tn = 0; tn < 2; ++tn) {
        const half_t* vs = vtp + ((size_t)wh * 32 + tn * 16 + (lane & 15)) * 256 +
                           kb + ((lane >> 4) << 3);
        v16h bv;
        uint4* bp = (uint4*)&bv;
        bp[0] = *(const uint4*)vs;
        bp[1] = *(const uint4*)(vs + 16);
        accO[tn] = wmma32(a, bv, accO[tn]);
      }
    }
    __syncthreads();
  }

  // ---- write O: o[win*256 + t][head*32 + hd] fp16 row-major ----
#pragma unroll
  for (int tn = 0; tn < 2; ++tn) {
#pragma unroll
    for (int r = 0; r < 8; ++r) {
      int t = qb * 128 + waveId * 16 + ((lane >> 4) << 3) + r;
      int c = head * 32 + tn * 16 + (lane & 15);
      o[((size_t)win * 256 + t) * 384 + c] = (half_t)accO[tn][r];
    }
  }
}

// ---------------------------------------------------------------------------
// Host-side launcher
// ---------------------------------------------------------------------------
extern "C" void kernel_launch(void* const* d_in, const int* in_sizes, int n_in,
                              void* d_out, int out_size, void* d_ws,
                              size_t ws_size, hipStream_t stream) {
  (void)in_sizes; (void)n_in; (void)out_size; (void)ws_size;
  const float* x     = (const float*)d_in[0];
  const float* n1w   = (const float*)d_in[1];
  const float* n1b   = (const float*)d_in[2];
  const float* qkvw  = (const float*)d_in[3];
  const float* qkvb  = (const float*)d_in[4];
  const float* rpb   = (const float*)d_in[5];
  const float* projw = (const float*)d_in[6];
  const float* projb = (const float*)d_in[7];
  const float* n2w   = (const float*)d_in[8];
  const float* n2b   = (const float*)d_in[9];
  const float* fc1w  = (const float*)d_in[10];
  const float* fc1b  = (const float*)d_in[11];
  const float* fc2w  = (const float*)d_in[12];
  const float* fc2b  = (const float*)d_in[13];
  float* out = (float*)d_out;
  char*  ws  = (char*)d_ws;

  // workspace layout (bytes)
  float*  bias12 = (float*)ws;                                  // 3,145,728
  half_t* wq = (half_t*)(ws + 3145728);                         //   884,736
  half_t* wp = (half_t*)(ws + 3145728 + 884736);                //   294,912
  half_t* w1 = (half_t*)(ws + 3145728 + 884736 + 294912);       // 1,179,648
  half_t* w2 = (half_t*)(ws + 3145728 + 884736 + 294912 + 1179648);
  const size_t R = 3145728 + 884736 + 294912 + 1179648 + 1179648; // 6,684,672
  half_t* hwin   = (half_t*)(ws + R);            // 32768x384 f16 (25,165,824B)
  half_t* qpack  = hwin + 12582912;              // 128*12 x 256 x 32
  half_t* kpack  = qpack + 12582912;
  half_t* vtpack = kpack + 12582912;
  half_t* a1     = hwin;                         // alias: fc1 out 32768x1536
  half_t* attno  = (half_t*)(ws + R + 4 * 25165824ULL);  // 32768x384 f16
  half_t* h2     = attno;                        // alias: LN2 out

  // 1) weight conversions to f16
  f32_to_f16_kernel<<<(442368 + 255) / 256, 256, 0, stream>>>(qkvw, wq, 442368);
  f32_to_f16_kernel<<<(147456 + 255) / 256, 256, 0, stream>>>(projw, wp, 147456);
  f32_to_f16_kernel<<<(589824 + 255) / 256, 256, 0, stream>>>(fc1w, w1, 589824);
  f32_to_f16_kernel<<<(589824 + 255) / 256, 256, 0, stream>>>(fc2w, w2, 589824);

  // 2) relative-position bias expansion
  bias_expand_kernel<<<3072, 256, 0, stream>>>(rpb, bias12);

  // 3) LN1 + cyclic shift + window partition -> f16
  ln_kernel<<<4096, 256, 0, stream>>>(x, n1w, n1b, hwin, 0);

  // 4) qkv GEMM (32768x1152x384) with q/k/vT packing epilogue
  wgemm_kernel<0><<<dim3(1152 / 64, 32768 / 128), 256, 0, stream>>>(
      hwin, wq, qkvb, 32768, 1152, 384, nullptr, qpack, kpack, vtpack, nullptr);

  // 5) windowed attention (128 win x 12 heads x 2 q-blocks)
  attn_kernel<<<128 * 12 * 2, 256, 0, stream>>>(qpack, kpack, vtpack, bias12,
                                                attno);

  // 6) proj GEMM + window reverse + unshift + residual -> d_out (fp32)
  wgemm_kernel<1><<<dim3(384 / 64, 32768 / 128), 256, 0, stream>>>(
      attno, wp, projb, 32768, 384, 384, out, nullptr, nullptr, nullptr, x);

  // 7) LN2 -> f16
  ln_kernel<<<4096, 256, 0, stream>>>(out, n2w, n2b, h2, 1);

  // 8) fc1 GEMM + exact GELU -> f16
  wgemm_kernel<2><<<dim3(1536 / 64, 32768 / 128), 256, 0, stream>>>(
      h2, w1, fc1b, 32768, 1536, 384, nullptr, a1, nullptr, nullptr, nullptr);

  // 9) fc2 GEMM, accumulate into d_out
  wgemm_kernel<3><<<dim3(6, 256), 256, 0, stream>>>(
      a1, w2, fc2b, 32768, 384, 1536, out, nullptr, nullptr, nullptr, nullptr);
}
